// CriticNetwork_52321291600334
// MI455X (gfx1250) — compile-verified, matrix operand order checked by reading, not run on hardware
//
#include <hip/hip_runtime.h>
#include <hip/hip_bf16.h>
#include <math.h>

// Dims from the reference
#define BDIM 256
#define SDIM 512
#define EDIM 128
#define HDIM 256
#define GDIM 1024  // 4*H

typedef _Float16 v16h __attribute__((ext_vector_type(16)));
typedef _Float16 v8h  __attribute__((ext_vector_type(8)));
typedef float    v8f  __attribute__((ext_vector_type(8)));
// GCC-style int4 to match the async-LDS builtin's parameter type exactly.
typedef int      gv4i __attribute__((vector_size(16)));

// Async global->LDS path (gfx1250). Guarded so both toolchains compile.
#if defined(__has_builtin)
#if __has_builtin(__builtin_amdgcn_global_load_async_to_lds_b128) && \
    __has_builtin(__builtin_amdgcn_s_wait_asynccnt)
#define HAVE_ASYNC_LDS 1
#endif
#endif
#ifndef HAVE_ASYNC_LDS
#define HAVE_ASYNC_LDS 0
#endif

// ---------------------------------------------------------------------------
// WMMA helper (wave32, 16x16x32 f16 -> f32)
// ---------------------------------------------------------------------------
__device__ __forceinline__ v8f wmma16(v16h a, v16h b, v8f c) {
  return __builtin_amdgcn_wmma_f32_16x16x32_f16(
      /*neg_a=*/false, a, /*neg_b=*/false, b,
      /*c_mod=*/(short)0, c, /*reuse_a=*/false, /*reuse_b=*/false);
}

// Load a 16x32 f16 tile in the documented A-operand layout.
// base: row-major [rows x ld] f16.  r0: first row, k0: first K (multiple of 32).
// lanes 0-15: row r0+lane, K = k0+0..7 and k0+16..23
// lanes16-31: row r0+lane-16, K = k0+8..15 and k0+24..31
// Serves the B operand of C = A * W^T too (W row-major [N x K], contiguous K).
// Works for both global and LDS base pointers (generic loads).
__device__ __forceinline__ v16h load_tile16(const _Float16* __restrict__ base,
                                            int ld, int r0, int k0, int lane) {
  int r  = r0 + (lane & 15);
  int kb = k0 + ((lane & 16) ? 8 : 0);
  const _Float16* p = base + (size_t)r * ld + kb;
  union { v16h v; v8h h[2]; } u;
  u.h[0] = *(const v8h*)p;
  u.h[1] = *(const v8h*)(p + 16);
  return u.v;
}

__device__ __forceinline__ float sigf(float x) {
  return 1.0f / (1.0f + __expf(-x));
}

// ---------------------------------------------------------------------------
// Prep: fold emb-GEMM into rank-2 Wfuse = enc_Wih @ W_emb; fold pb input
// projection into constant vector xp; convert all recurrent weights to f16.
// ---------------------------------------------------------------------------
__global__ void prep_kernel(const float* __restrict__ encWhh,
                            const float* __restrict__ pbWhh,
                            const float* __restrict__ Wq,
                            const float* __restrict__ Wref,
                            const float* __restrict__ Wd1,
                            const float* __restrict__ encWih,
                            const float* __restrict__ Wemb,
                            const float* __restrict__ encbih,
                            const float* __restrict__ encbhh,
                            const float* __restrict__ pbWih,
                            const float* __restrict__ dec,
                            const float* __restrict__ pbbih,
                            const float* __restrict__ pbbhh,
                            _Float16* __restrict__ WhhE,
                            _Float16* __restrict__ WhhP,
                            _Float16* __restrict__ WqH,
                            _Float16* __restrict__ WrefH,
                            _Float16* __restrict__ Wd1H,
                            float* __restrict__ Wfuse,
                            float* __restrict__ xbE,
                            float* __restrict__ xp) {
  int i = blockIdx.x * blockDim.x + threadIdx.x;
  if (i < GDIM * HDIM) {
    WhhE[i] = (_Float16)encWhh[i];
    WhhP[i] = (_Float16)pbWhh[i];
  }
  if (i < HDIM * HDIM) {
    WqH[i]   = (_Float16)Wq[i];
    WrefH[i] = (_Float16)Wref[i];
    Wd1H[i]  = (_Float16)Wd1[i];
  }
  if (i < 2 * GDIM) {  // Wfuse[j][c] = sum_e enc_Wih[j,e] * W_emb[e,c]
    int j = i >> 1, c = i & 1;
    float acc = 0.f;
    for (int e = 0; e < EDIM; ++e) acc += encWih[j * EDIM + e] * Wemb[e * 2 + c];
    Wfuse[i] = acc;
  }
  if (i < GDIM) {
    xbE[i] = encbih[i] + encbhh[i];
    float a = 0.f;
    for (int e = 0; e < EDIM; ++e) a += pbWih[i * EDIM + e] * dec[e];
    xp[i] = a + pbbih[i] + pbbhh[i];
  }
}

__global__ void zero_state(_Float16* __restrict__ h0, _Float16* __restrict__ h1,
                           float* __restrict__ c) {
  int i = blockIdx.x * blockDim.x + threadIdx.x;
  if (i < BDIM * HDIM) { h0[i] = (_Float16)0.f; h1[i] = (_Float16)0.f; c[i] = 0.f; }
}

// ---------------------------------------------------------------------------
// One LSTM timestep. grid = (16 n-tiles, 16 m-tiles), block = 32 (one wave).
// Each wave owns tile (m0..m0+15, n0..n0+15) of ALL FOUR gates so the
// elementwise c/h update happens entirely in the WMMA accumulator registers.
// gates = x_proj + h @ Whh^T, with x_proj either rank-2 (encoder) or a
// broadcast vector (process block), folded into the initial C matrix.
// Inner loop is software-pipelined: tiles for k-step kk+1 are issued before
// the WMMAs of k-step kk, overlapping VMEM latency with matrix math.
// ---------------------------------------------------------------------------
__global__ void lstm_step(const _Float16* __restrict__ hin,   // [B,H] f16
                          float* __restrict__ cbuf,           // [B,H] f32 (in/out)
                          const _Float16* __restrict__ Whh,   // [4H,H] f16
                          const float* __restrict__ xb,       // [4H] bias (enc) or xp (pb)
                          const float* __restrict__ Wfuse,    // [4H,2] or nullptr
                          const float* __restrict__ inp_s,    // &inp[0][s][0] or nullptr
                          _Float16* __restrict__ hout,        // [B,H] f16
                          _Float16* __restrict__ enc_out_s) { // &enc_out[0][s][0] or nullptr
  int lane = threadIdx.x;
  int n0 = blockIdx.x * 16;
  int m0 = blockIdx.y * 16;
  int colL = n0 + (lane & 15);
  int rhi = (lane & 16) ? 8 : 0;

  v8f acc[4];
  if (Wfuse) {
#pragma unroll
    for (int q = 0; q < 4; ++q) {
      int j = q * HDIM + colL;
      float w0 = Wfuse[j * 2], w1 = Wfuse[j * 2 + 1], bb = xb[j];
#pragma unroll
      for (int r = 0; r < 8; ++r) {
        int row = m0 + r + rhi;
        float x0 = inp_s[(size_t)row * (SDIM * 2)];
        float x1 = inp_s[(size_t)row * (SDIM * 2) + 1];
        acc[q][r] = fmaf(x0, w0, fmaf(x1, w1, bb));
      }
    }
  } else {
#pragma unroll
    for (int q = 0; q < 4; ++q) {
      float bb = xb[q * HDIM + colL];
#pragma unroll
      for (int r = 0; r < 8; ++r) acc[q][r] = bb;
    }
  }

  const int KK = HDIM / 32;  // 8
  v16h a_cur = load_tile16(hin, HDIM, m0, 0, lane);
  v16h b0 = load_tile16(Whh, HDIM, 0 * HDIM + n0, 0, lane);
  v16h b1 = load_tile16(Whh, HDIM, 1 * HDIM + n0, 0, lane);
  v16h b2 = load_tile16(Whh, HDIM, 2 * HDIM + n0, 0, lane);
  v16h b3 = load_tile16(Whh, HDIM, 3 * HDIM + n0, 0, lane);
#pragma unroll
  for (int kk = 0; kk < KK; ++kk) {
    v16h a_n, c0, c1, c2, c3;
    if (kk + 1 < KK) {  // prefetch next k-step before consuming current
      int k1 = (kk + 1) * 32;
      a_n = load_tile16(hin, HDIM, m0, k1, lane);
      c0 = load_tile16(Whh, HDIM, 0 * HDIM + n0, k1, lane);
      c1 = load_tile16(Whh, HDIM, 1 * HDIM + n0, k1, lane);
      c2 = load_tile16(Whh, HDIM, 2 * HDIM + n0, k1, lane);
      c3 = load_tile16(Whh, HDIM, 3 * HDIM + n0, k1, lane);
    }
    acc[0] = wmma16(a_cur, b0, acc[0]);
    acc[1] = wmma16(a_cur, b1, acc[1]);
    acc[2] = wmma16(a_cur, b2, acc[2]);
    acc[3] = wmma16(a_cur, b3, acc[3]);
    if (kk + 1 < KK) { a_cur = a_n; b0 = c0; b1 = c1; b2 = c2; b3 = c3; }
  }

#pragma unroll
  for (int r = 0; r < 8; ++r) {
    int row = m0 + r + rhi;
    size_t idx = (size_t)row * HDIM + colL;
    float ig = sigf(acc[0][r]);
    float fg = sigf(acc[1][r]);
    float gg = tanhf(acc[2][r]);
    float og = sigf(acc[3][r]);
    float cn = fg * cbuf[idx] + ig * gg;
    float hn = og * tanhf(cn);
    cbuf[idx] = cn;
    hout[idx] = (_Float16)hn;
    if (enc_out_s) enc_out_s[(size_t)row * (SDIM * HDIM) + colL] = (_Float16)hn;
  }
}

// ---------------------------------------------------------------------------
// Generic small GEMM: C = [relu](A @ W^T + bias).  grid=(N/16, M/16), block=32.
// Used for q = h@Wq^T+bq and t1 = relu(g@Wd1^T).  Software-pipelined.
// ---------------------------------------------------------------------------
__global__ void gemm_nt_16(const _Float16* __restrict__ A,
                           const _Float16* __restrict__ W,
                           const float* __restrict__ bias,
                           float* __restrict__ Cout, int K, int relu) {
  int lane = threadIdx.x;
  int n0 = blockIdx.x * 16;
  int m0 = blockIdx.y * 16;
  int N = gridDim.x * 16;
  int colL = n0 + (lane & 15);
  int rhi = (lane & 16) ? 8 : 0;
  v8f acc;
  float bb = bias ? bias[colL] : 0.0f;
#pragma unroll
  for (int r = 0; r < 8; ++r) acc[r] = bb;
  int nk = K / 32;
  v16h a_cur = load_tile16(A, K, m0, 0, lane);
  v16h b_cur = load_tile16(W, K, n0, 0, lane);
  for (int kk = 0; kk < nk; ++kk) {
    v16h a_n, b_n;
    if (kk + 1 < nk) {
      a_n = load_tile16(A, K, m0, (kk + 1) * 32, lane);
      b_n = load_tile16(W, K, n0, (kk + 1) * 32, lane);
    }
    acc = wmma16(a_cur, b_cur, acc);
    if (kk + 1 < nk) { a_cur = a_n; b_cur = b_n; }
  }
#pragma unroll
  for (int r = 0; r < 8; ++r) {
    int row = m0 + r + rhi;
    float v = acc[r];
    if (relu) v = fmaxf(v, 0.0f);
    Cout[(size_t)row * N + colL] = v;
  }
}

// ---------------------------------------------------------------------------
// u[row] = V . tanh(enc[row,:] @ Wref^T + bref + q[b,:])   (row = b*S + s)
// grid = B*S/16 blocks, 256 threads (8 waves). The 16x256 A-stripe is shared
// by all 8 waves -> stage it once per block into LDS via the CDNA5 async
// global->LDS path (ASYNCcnt), then each wave runs 2 N-tiles with B tiles
// prefetch-pipelined from global, and reduces across H via LDS atomics.
// ---------------------------------------------------------------------------
__global__ void attn_u_kernel(const _Float16* __restrict__ enc,   // [B*S,H] f16
                              const _Float16* __restrict__ WrefH, // [H,H] f16
                              const float* __restrict__ brefv,    // [H]
                              const float* __restrict__ qv,       // [B,H]
                              const float* __restrict__ Vv,       // [H]
                              float* __restrict__ u) {            // [B*S]
  __shared__ __attribute__((aligned(16))) _Float16 Atile[16 * HDIM];  // 8 KB
  __shared__ float rowsum[16];
  int tid = threadIdx.x;
  int lane = tid & 31;
  int wave = tid >> 5;
  int m0 = blockIdx.x * 16;
  int rhi = (lane & 16) ? 8 : 0;

  // ---- stage A stripe (16 rows x 256 cols f16 = 8192 B) into LDS ----
  {
    const _Float16* gsrc = enc + (size_t)m0 * HDIM;
#if HAVE_ASYNC_LDS
#pragma unroll
    for (int j = 0; j < 2; ++j) {
      int e = (tid * 2 + j) * 8;  // 512 chunks of 8 halves (16 B)
      gv4i* gp = (gv4i*)const_cast<_Float16*>(gsrc + e);
      gv4i* lp = (gv4i*)&Atile[e];
      __builtin_amdgcn_global_load_async_to_lds_b128(
          (__attribute__((address_space(1))) gv4i*)gp,
          (__attribute__((address_space(3))) gv4i*)lp,
          /*offset=*/0, /*cpol=*/0);
    }
    __builtin_amdgcn_s_wait_asynccnt(0);
#else
    const v8h* gs = (const v8h*)gsrc;
    v8h* ld = (v8h*)Atile;
#pragma unroll
    for (int j = 0; j < 2; ++j) ld[tid * 2 + j] = gs[tid * 2 + j];
#endif
  }
  if (tid < 16) rowsum[tid] = 0.0f;
  __syncthreads();

#pragma unroll
  for (int t = 0; t < 2; ++t) {
    int n0 = (wave * 2 + t) * 16;
    int colL = n0 + (lane & 15);
    v8f acc;
#pragma unroll
    for (int r = 0; r < 8; ++r) {
      int row = m0 + r + rhi;
      int b = row >> 9;  // row / S
      acc[r] = brefv[colL] + qv[(size_t)b * HDIM + colL];
    }
    v16h b_cur = load_tile16(WrefH, HDIM, n0, 0, lane);
#pragma unroll
    for (int kk = 0; kk < HDIM / 32; ++kk) {
      v16h a = load_tile16(Atile, HDIM, 0, kk * 32, lane);  // LDS (ds loads)
      v16h b_n;
      if (kk + 1 < HDIM / 32)
        b_n = load_tile16(WrefH, HDIM, n0, (kk + 1) * 32, lane);
      acc = wmma16(a, b_cur, acc);
      if (kk + 1 < HDIM / 32) b_cur = b_n;
    }
    float vcol = Vv[colL];
#pragma unroll
    for (int r = 0; r < 8; ++r) {
      atomicAdd(&rowsum[r + rhi], tanhf(acc[r]) * vcol);
    }
  }
  __syncthreads();
  if (tid < 16) u[m0 + tid] = rowsum[tid];
}

// ---------------------------------------------------------------------------
// Per-batch softmax over S, then glimpse g[b,h] = sum_s align[s]*enc[b,s,h].
// grid = B blocks, 256 threads.
// ---------------------------------------------------------------------------
__global__ void softmax_glimpse(const float* __restrict__ u,
                                const _Float16* __restrict__ enc,
                                _Float16* __restrict__ gf16) {
  __shared__ float su[SDIM];
  __shared__ float red[256];
  int b = blockIdx.x, tid = threadIdx.x;
  float a0 = u[(size_t)b * SDIM + tid];
  float a1 = u[(size_t)b * SDIM + 256 + tid];
  red[tid] = fmaxf(a0, a1);
  __syncthreads();
  for (int sft = 128; sft > 0; sft >>= 1) {
    if (tid < sft) red[tid] = fmaxf(red[tid], red[tid + sft]);
    __syncthreads();
  }
  float mx = red[0];
  __syncthreads();
  float e0 = __expf(a0 - mx), e1 = __expf(a1 - mx);
  su[tid] = e0;
  su[tid + 256] = e1;
  red[tid] = e0 + e1;
  __syncthreads();
  for (int sft = 128; sft > 0; sft >>= 1) {
    if (tid < sft) red[tid] += red[tid + sft];
    __syncthreads();
  }
  float inv = 1.0f / red[0];
  __syncthreads();
  // glimpse: one output column per thread
  float acc = 0.0f;
  const _Float16* ep = enc + (size_t)b * SDIM * HDIM + tid;
  for (int ss = 0; ss < SDIM; ++ss) acc += su[ss] * (float)ep[(size_t)ss * HDIM];
  gf16[(size_t)b * HDIM + tid] = (_Float16)(acc * inv);
}

// out[b] = t1[b,:] . Wd2  (final head reduction)
__global__ void head_out(const float* __restrict__ t1,
                         const float* __restrict__ Wd2,
                         float* __restrict__ out) {
  __shared__ float red[256];
  int b = blockIdx.x, tid = threadIdx.x;
  red[tid] = t1[(size_t)b * HDIM + tid] * Wd2[tid];
  __syncthreads();
  for (int sft = 128; sft > 0; sft >>= 1) {
    if (tid < sft) red[tid] += red[tid + sft];
    __syncthreads();
  }
  if (tid == 0) out[b] = red[0];
}

// ---------------------------------------------------------------------------
// Host side
// ---------------------------------------------------------------------------
extern "C" void kernel_launch(void* const* d_in, const int* in_sizes, int n_in,
                              void* d_out, int out_size, void* d_ws, size_t ws_size,
                              hipStream_t stream) {
  (void)in_sizes; (void)n_in; (void)out_size; (void)ws_size;
  const float* inp      = (const float*)d_in[0];
  const float* W_emb    = (const float*)d_in[1];
  const float* dec_in   = (const float*)d_in[2];
  const float* enc_Wih  = (const float*)d_in[3];
  const float* enc_Whh  = (const float*)d_in[4];
  const float* enc_bih  = (const float*)d_in[5];
  const float* enc_bhh  = (const float*)d_in[6];
  const float* pb_Wih   = (const float*)d_in[7];
  const float* pb_Whh   = (const float*)d_in[8];
  const float* pb_bih   = (const float*)d_in[9];
  const float* pb_bhh   = (const float*)d_in[10];
  const float* Wq       = (const float*)d_in[11];
  const float* bq       = (const float*)d_in[12];
  const float* Wref     = (const float*)d_in[13];
  const float* bref     = (const float*)d_in[14];
  const float* V        = (const float*)d_in[15];
  const float* Wd1      = (const float*)d_in[16];
  const float* Wd2      = (const float*)d_in[17];

  char* ws = (char*)d_ws;
  size_t off = 0;
  auto alloc = [&](size_t bytes) -> void* {
    void* p = ws + off;
    off = (off + bytes + 255) & ~(size_t)255;
    return p;
  };
  _Float16* WhhE  = (_Float16*)alloc((size_t)GDIM * HDIM * 2);
  _Float16* WhhP  = (_Float16*)alloc((size_t)GDIM * HDIM * 2);
  _Float16* WqH   = (_Float16*)alloc((size_t)HDIM * HDIM * 2);
  _Float16* WrefH = (_Float16*)alloc((size_t)HDIM * HDIM * 2);
  _Float16* Wd1H  = (_Float16*)alloc((size_t)HDIM * HDIM * 2);
  float*    Wfuse = (float*)alloc((size_t)GDIM * 2 * 4);
  float*    xbE   = (float*)alloc((size_t)GDIM * 4);
  float*    xp    = (float*)alloc((size_t)GDIM * 4);
  _Float16* hb0   = (_Float16*)alloc((size_t)BDIM * HDIM * 2);
  _Float16* hb1   = (_Float16*)alloc((size_t)BDIM * HDIM * 2);
  float*    cb    = (float*)alloc((size_t)BDIM * HDIM * 4);
  float*    qbuf  = (float*)alloc((size_t)BDIM * HDIM * 4);
  _Float16* encO  = (_Float16*)alloc((size_t)BDIM * SDIM * HDIM * 2);  // 64 MB
  float*    ubuf  = (float*)alloc((size_t)BDIM * SDIM * 4);
  _Float16* gf16  = (_Float16*)alloc((size_t)BDIM * HDIM * 2);
  float*    t1    = (float*)alloc((size_t)BDIM * HDIM * 4);

  prep_kernel<<<(GDIM * HDIM + 255) / 256, 256, 0, stream>>>(
      enc_Whh, pb_Whh, Wq, Wref, Wd1, enc_Wih, W_emb, enc_bih, enc_bhh,
      pb_Wih, dec_in, pb_bih, pb_bhh,
      WhhE, WhhP, WqH, WrefH, Wd1H, Wfuse, xbE, xp);
  zero_state<<<(BDIM * HDIM + 255) / 256, 256, 0, stream>>>(hb0, hb1, cb);

  _Float16* hbuf[2] = {hb0, hb1};
  int par = 0;
  // encoder scan
  for (int s = 0; s < SDIM; ++s) {
    lstm_step<<<dim3(16, 16), 32, 0, stream>>>(
        hbuf[par], cb, WhhE, xbE, Wfuse, inp + (size_t)s * 2,
        hbuf[par ^ 1], encO + (size_t)s * HDIM);
    par ^= 1;
  }
  // process block scan (constant input, no per-step output)
  for (int t = 0; t < SDIM; ++t) {
    lstm_step<<<dim3(16, 16), 32, 0, stream>>>(
        hbuf[par], cb, WhhP, xp, nullptr, nullptr, hbuf[par ^ 1], nullptr);
    par ^= 1;
  }

  // q = h @ Wq^T + bq
  gemm_nt_16<<<dim3(HDIM / 16, BDIM / 16), 32, 0, stream>>>(
      hbuf[par], WqH, bq, qbuf, HDIM, 0);
  // u = V . tanh(enc@Wref^T + bref + q)
  attn_u_kernel<<<(BDIM * SDIM) / 16, 256, 0, stream>>>(
      encO, WrefH, bref, qbuf, V, ubuf);
  // softmax + glimpse
  softmax_glimpse<<<BDIM, 256, 0, stream>>>(ubuf, encO, gf16);
  // t1 = relu(g @ Wd1^T)
  gemm_nt_16<<<dim3(HDIM / 16, BDIM / 16), 32, 0, stream>>>(
      gf16, Wd1H, nullptr, t1, HDIM, 1);
  // out = t1 @ Wd2^T
  head_out<<<BDIM, 256, 0, stream>>>(t1, Wd2, (float*)d_out);
}
